// MTGCN_9294309229063
// MI455X (gfx1250) — matrix operand order repaired and checked by reading, not compiled, not run on hardware
//
#include <hip/hip_runtime.h>
#include <hip/hip_bf16.h>

typedef __attribute__((ext_vector_type(16))) _Float16 v16h;
typedef __attribute__((ext_vector_type(8)))  _Float16 v8h;
typedef __attribute__((ext_vector_type(8)))  float    v8f;
typedef __attribute__((ext_vector_type(4)))  float    f4;

// ---------------------------------------------------------------------------
// Edge preprocessing
// ---------------------------------------------------------------------------
__global__ void __launch_bounds__(256) deg_kernel(const int* __restrict__ src,
                                                  float* __restrict__ deg, int E) {
    int e = blockIdx.x * blockDim.x + threadIdx.x;
    if (e < E) atomicAdd(&deg[src[e]], 1.0f);
}

__global__ void __launch_bounds__(256) norm_kernel(const int* __restrict__ src,
                                                   const int* __restrict__ dst,
                                                   const float* __restrict__ deg,
                                                   float* __restrict__ nrm, int E) {
    int e = blockIdx.x * blockDim.x + threadIdx.x;
    if (e >= E) return;
    float a = deg[src[e]], b = deg[dst[e]];
    float da = (a > 0.f) ? rsqrtf(a) : 0.f;
    float db = (b > 0.f) ? rsqrtf(b) : 0.f;
    nrm[e] = -(da * db);
}

// tx[dst[e], f] += X[src[e], f] * norm[e]; 2 features per thread (F is even)
__global__ void __launch_bounds__(256) scatter2_kernel(const float* __restrict__ X,
                                                       const int* __restrict__ src,
                                                       const int* __restrict__ dst,
                                                       const float* __restrict__ nrm,
                                                       float* __restrict__ outp,
                                                       long long E, int F) {
    const int Fh = F >> 1;
    long long t = (long long)blockIdx.x * blockDim.x + threadIdx.x;
    long long total = E * (long long)Fh;
    if (t >= total) return;
    long long e = t / Fh;
    int f = (int)(t - e * (long long)Fh) * 2;
    int s = src[e], d = dst[e];
    float w = nrm[e];
    const float* xr = X    + (long long)s * F + f;
    float*       orow = outp + (long long)d * F + f;
    atomicAdd(orow,     xr[0] * w);
    atomicAdd(orow + 1, xr[1] * w);
}

// ---------------------------------------------------------------------------
// Operand packing for WMMA (fp32 -> f16, zero-padded)
// ---------------------------------------------------------------------------
// A[M,K] fp32 row-major -> O[M,Kp] f16 (Kp multiple of 32)
__global__ void __launch_bounds__(256) pack_a16(const float* __restrict__ A,
                                                _Float16* __restrict__ O,
                                                long long M, int K, int Kp) {
    long long t = (long long)blockIdx.x * blockDim.x + threadIdx.x;
    long long total = M * (long long)Kp;
    if (t >= total) return;
    long long m = t / Kp;
    int k = (int)(t - m * (long long)Kp);
    O[t] = (k < K) ? (_Float16)A[m * (long long)K + k] : (_Float16)0.f;
}

// B element (k,n) = B[k*brs + n*bcs]  ->  O[n*Kp + k] f16  (Np x Kp, padded)
__global__ void __launch_bounds__(256) pack_b16t(const float* __restrict__ B,
                                                 _Float16* __restrict__ O,
                                                 int K, int Nc, int brs, int bcs,
                                                 int Kp, int Np) {
    int t = blockIdx.x * blockDim.x + threadIdx.x;
    int total = Np * Kp;
    if (t >= total) return;
    int n = t / Kp, k = t - n * Kp;
    O[t] = (k < K && n < Nc)
             ? (_Float16)B[(long long)k * brs + (long long)n * bcs]
             : (_Float16)0.f;
}

// ---------------------------------------------------------------------------
// Packed-f16 WMMA GEMM:  Out = [addend +] relu_opt( A1@B1 [+ A2@B2] + bias )
// A*: [M,Kp] f16 row-major.  B*: [Np,Kp] f16 (transposed/packed: Bt[n][k]).
// One wave32 -> one 16x16 f32 tile via V_WMMA_F32_16X16X32_F16.
// All fragment loads are unconditional contiguous 16-byte vector loads.
// ---------------------------------------------------------------------------
__device__ __forceinline__ v16h load_afrag(const _Float16* __restrict__ arow,
                                           int k0, int half) {
    // lane half h: elements 0..7 = K k0+8h.., elements 8..15 = K k0+16+8h..
    v8h lo = *(const v8h*)(arow + k0 + 8 * half);
    v8h hi = *(const v8h*)(arow + k0 + 16 + 8 * half);
    v16h r;
#pragma unroll
    for (int i = 0; i < 8; ++i) { r[i] = lo[i]; r[i + 8] = hi[i]; }
    return r;
}

__device__ __forceinline__ v16h load_bfrag(const _Float16* __restrict__ brow,
                                           int k0, int half) {
    // lane half h: elements 0..7 = K k0+8h.., elements 8..15 = K k0+16+8h..
    v8h lo = *(const v8h*)(brow + k0 + 8 * half);
    v8h hi = *(const v8h*)(brow + k0 + 16 + 8 * half);
    v16h r;
#pragma unroll
    for (int i = 0; i < 8; ++i) { r[i] = lo[i]; r[i + 8] = hi[i]; }
    return r;
}

__global__ void __launch_bounds__(128) gemm_wmma_packed(
    const _Float16* __restrict__ A1, const _Float16* __restrict__ B1,
    const _Float16* __restrict__ A2, const _Float16* __restrict__ B2,
    const float* __restrict__ bias, const float* __restrict__ addend,
    float* __restrict__ Out, int M, int Kp, int Nc, int do_relu)
{
    const int lane   = threadIdx.x & 31;
    const int tilesN = (Nc + 15) >> 4;
    const int tilesM = (M + 15) >> 4;
    const int tile   = blockIdx.x * 4 + (threadIdx.x >> 5);
    if (tile >= tilesM * tilesN) return;          // wave-uniform exit
    const int tm = tile / tilesN;
    const int tn = tile % tilesN;

    const int half = lane >> 4;
    const int mr   = lane & 15;

    const _Float16* arow1 = A1 + (long long)(tm * 16 + mr) * Kp;
    const _Float16* brow1 = B1 + (long long)(tn * 16 + mr) * Kp;

    v8f acc = {0.f, 0.f, 0.f, 0.f, 0.f, 0.f, 0.f, 0.f};

    for (int k0 = 0; k0 < Kp; k0 += 32) {
        acc = __builtin_amdgcn_wmma_f32_16x16x32_f16(
                  false, load_afrag(arow1, k0, half),
                  false, load_bfrag(brow1, k0, half),
                  (short)0, acc, false, false);
    }
    if (A2) {                                     // uniform branch (Cheb second term)
        const _Float16* arow2 = A2 + (long long)(tm * 16 + mr) * Kp;
        const _Float16* brow2 = B2 + (long long)(tn * 16 + mr) * Kp;
        for (int k0 = 0; k0 < Kp; k0 += 32) {
            acc = __builtin_amdgcn_wmma_f32_16x16x32_f16(
                      false, load_afrag(arow2, k0, half),
                      false, load_bfrag(brow2, k0, half),
                      (short)0, acc, false, false);
        }
    }

#pragma unroll
    for (int r = 0; r < 8; ++r) {                 // C/D: M = r + 8*half, N = lane&15
        int m = tm * 16 + r + 8 * half;
        int n = tn * 16 + (lane & 15);
        if (m < M && n < Nc) {
            float v = acc[r];
            if (bias)               v += bias[n];
            if (do_relu && v < 0.f) v = 0.f;
            if (addend)             v += addend[(long long)m * Nc + n];
            Out[(long long)m * Nc + n] = v;
        }
    }
}

// ---------------------------------------------------------------------------
// Link-prediction loss (float4 dot; F divisible by 4, rows 16B-aligned)
// ---------------------------------------------------------------------------
__global__ void __launch_bounds__(256) edge_loss_kernel(
    const float* __restrict__ z, const int* __restrict__ ia,
    const int* __restrict__ ib, int E, int F, int neg,
    float* __restrict__ accum)
{
    __shared__ float sbuf[256];
    int e = blockIdx.x * blockDim.x + threadIdx.x;
    float term = 0.f;
    if (e < E) {
        const f4* za = (const f4*)(z + (long long)ia[e] * F);
        const f4* zb = (const f4*)(z + (long long)ib[e] * F);
        float s = 0.f;
        for (int f = 0; f < (F >> 2); ++f) {
            f4 a = za[f], b = zb[f];
            s += a[0] * b[0] + a[1] * b[1] + a[2] * b[2] + a[3] * b[3];
        }
        float sig = 1.f / (1.f + __expf(-s));
        term = neg ? __logf(1.f - sig + 1e-15f) : __logf(sig + 1e-15f);
    }
    sbuf[threadIdx.x] = term;
    __syncthreads();
    for (int off = 128; off > 0; off >>= 1) {
        if ((int)threadIdx.x < off) sbuf[threadIdx.x] += sbuf[threadIdx.x + off];
        __syncthreads();
    }
    if (threadIdx.x == 0) atomicAdd(&accum[neg], sbuf[0]);
}

__global__ void finalize_kernel(const float* __restrict__ accum,
                                const float* __restrict__ c1,
                                const float* __restrict__ c2,
                                float* __restrict__ outv, int E)
{
    if (blockIdx.x == 0 && threadIdx.x == 0) {
        outv[0] = -(accum[0] / (float)E) - (accum[1] / (float)E);
        outv[1] = c1[0];
        outv[2] = c2[0];
    }
}

// Output layer (Nc=1): out[i] = b3 + sum_f xm[i,f]*W3_0[f] + txm[i,f]*W3_1[f]
__global__ void __launch_bounds__(256) out3_kernel(
    const float* __restrict__ xm, const float* __restrict__ txm,
    const float* __restrict__ w0, const float* __restrict__ w1,
    const float* __restrict__ b3, float* __restrict__ out, int Nn, int F)
{
    int i = blockIdx.x * blockDim.x + threadIdx.x;
    if (i >= Nn) return;
    const f4* xr = (const f4*)(xm  + (long long)i * F);
    const f4* tr = (const f4*)(txm + (long long)i * F);
    const f4* w0v = (const f4*)w0;
    const f4* w1v = (const f4*)w1;
    float s = b3[0];
    for (int f = 0; f < (F >> 2); ++f) {
        f4 a = xr[f], b = tr[f], wa = w0v[f], wb = w1v[f];
        s += a[0] * wa[0] + a[1] * wa[1] + a[2] * wa[2] + a[3] * wa[3]
           + b[0] * wb[0] + b[1] * wb[1] + b[2] * wb[2] + b[3] * wb[3];
    }
    out[i] = s;
}

// ---------------------------------------------------------------------------
extern "C" void kernel_launch(void* const* d_in, const int* in_sizes, int n_in,
                              void* d_out, int out_size, void* d_ws, size_t ws_size,
                              hipStream_t stream)
{
    const float* x      = (const float*)d_in[0];
    const int*   ei     = (const int*)  d_in[1];
    const int*   nei    = (const int*)  d_in[2];
    const float* W1_0   = (const float*)d_in[3];
    const float* W1_1   = (const float*)d_in[4];
    const float* b1     = (const float*)d_in[5];
    const float* W2_0   = (const float*)d_in[6];
    const float* W2_1   = (const float*)d_in[7];
    const float* b2     = (const float*)d_in[8];
    const float* W3_0   = (const float*)d_in[9];
    const float* W3_1   = (const float*)d_in[10];
    const float* b3     = (const float*)d_in[11];
    const float* lin1_W = (const float*)d_in[12];
    const float* lin1_b = (const float*)d_in[13];
    const float* lin2_W = (const float*)d_in[14];
    const float* lin2_b = (const float*)d_in[15];
    const float* c1     = (const float*)d_in[16];
    const float* c2     = (const float*)d_in[17];

    const int Nn = in_sizes[0] / 58;   // 50000
    const int Ee = in_sizes[1] / 2;    // 800000
    const int* src = ei;
    const int* dst = ei + Ee;

    // ---- workspace layout ----
    float* ws   = (float*)d_ws;
    float* deg  = ws;                                    // N
    float* nrm  = deg + Nn;                              // E
    float* tx1  = nrm + Ee;                              // N*58   (fp32 scatter acc)
    float* h    = tx1 + (long long)Nn * 58;              // N*300
    float* th   = h   + (long long)Nn * 300;             // N*300  (fp32 scatter acc)
    float* x1   = th  + (long long)Nn * 300;             // N*100
    float* xm   = x1  + (long long)Nn * 100;             // N*100
    float* z    = xm  + (long long)Nn * 100;             // N*100
    float* txm  = z   + (long long)Nn * 100;             // N*100  (fp32 scatter acc)
    float* accl = txm + (long long)Nn * 100;             // 2

    _Float16* fp    = (_Float16*)(accl + 2);
    _Float16* x16   = fp;                                // N*64
    _Float16* tx16  = x16  + (long long)Nn * 64;         // N*64
    _Float16* h16   = tx16 + (long long)Nn * 64;         // N*320
    _Float16* th16  = h16  + (long long)Nn * 320;        // N*320
    _Float16* W10t  = th16 + (long long)Nn * 320;        // 304*64
    _Float16* W11t  = W10t + 304 * 64;                   // 304*64
    _Float16* W20t  = W11t + 304 * 64;                   // 112*320
    _Float16* W21t  = W20t + 112 * 320;                  // 112*320
    _Float16* L1t   = W21t + 112 * 320;                  // 112*64
    _Float16* L2t   = L1t  + 112 * 64;                   // 112*64

    hipMemsetAsync(deg,  0, sizeof(float) * (size_t)Nn,       stream);
    hipMemsetAsync(tx1,  0, sizeof(float) * (size_t)Nn * 58,  stream);
    hipMemsetAsync(th,   0, sizeof(float) * (size_t)Nn * 300, stream);
    hipMemsetAsync(txm,  0, sizeof(float) * (size_t)Nn * 100, stream);
    hipMemsetAsync(accl, 0, sizeof(float) * 2,                stream);

    const int T = 256;
    auto GB = [](long long n, int t) { return (int)((n + t - 1) / t); };

    deg_kernel <<<GB(Ee, T), T, 0, stream>>>(src, deg, Ee);
    norm_kernel<<<GB(Ee, T), T, 0, stream>>>(src, dst, deg, nrm, Ee);

    // weight + x packs (independent of edge work)
    pack_a16 <<<GB((long long)Nn * 64, T), T, 0, stream>>>(x, x16, Nn, 58, 64);
    pack_b16t<<<GB(304 * 64,  T), T, 0, stream>>>(W1_0,  W10t, 58, 300, 300, 1, 64, 304);
    pack_b16t<<<GB(304 * 64,  T), T, 0, stream>>>(W1_1,  W11t, 58, 300, 300, 1, 64, 304);
    pack_b16t<<<GB(112 * 320, T), T, 0, stream>>>(W2_0,  W20t, 300, 100, 100, 1, 320, 112);
    pack_b16t<<<GB(112 * 320, T), T, 0, stream>>>(W2_1,  W21t, 300, 100, 100, 1, 320, 112);
    pack_b16t<<<GB(112 * 64,  T), T, 0, stream>>>(lin1_W, L1t, 58, 100, 1, 58, 64, 112);
    pack_b16t<<<GB(112 * 64,  T), T, 0, stream>>>(lin2_W, L2t, 58, 100, 1, 58, 64, 112);

    // layer 1: tx1 = scatter(x); h = relu(x@W1_0 + tx1@W1_1 + b1)
    scatter2_kernel<<<GB((long long)Ee * 29, T), T, 0, stream>>>(x, src, dst, nrm, tx1, Ee, 58);
    pack_a16<<<GB((long long)Nn * 64, T), T, 0, stream>>>(tx1, tx16, Nn, 58, 64);
    {
        int tiles = ((Nn + 15) / 16) * ((300 + 15) / 16);
        gemm_wmma_packed<<<(tiles + 3) / 4, 128, 0, stream>>>(
            x16, W10t, tx16, W11t, b1, nullptr, h, Nn, 64, 300, 1);
    }

    // layer 2: th = scatter(h); x1 = relu(h@W2_0 + th@W2_1 + b2)
    scatter2_kernel<<<GB((long long)Ee * 150, T), T, 0, stream>>>(h, src, dst, nrm, th, Ee, 300);
    pack_a16<<<GB((long long)Nn * 320, T), T, 0, stream>>>(h,  h16,  Nn, 300, 320);
    pack_a16<<<GB((long long)Nn * 320, T), T, 0, stream>>>(th, th16, Nn, 300, 320);
    {
        int tiles = ((Nn + 15) / 16) * ((100 + 15) / 16);
        gemm_wmma_packed<<<(tiles + 3) / 4, 128, 0, stream>>>(
            h16, W20t, th16, W21t, b2, nullptr, x1, Nn, 320, 100, 1);
        // xm = x1 + relu(x@lin1^T + b); z = x1 + relu(x@lin2^T + b)
        gemm_wmma_packed<<<(tiles + 3) / 4, 128, 0, stream>>>(
            x16, L1t, nullptr, nullptr, lin1_b, x1, xm, Nn, 64, 100, 1);
        gemm_wmma_packed<<<(tiles + 3) / 4, 128, 0, stream>>>(
            x16, L2t, nullptr, nullptr, lin2_b, x1, z, Nn, 64, 100, 1);
    }

    // link-prediction loss
    edge_loss_kernel<<<GB(Ee, T), T, 0, stream>>>(z, ei,  ei  + Ee, Ee, 100, 0, accl);
    edge_loss_kernel<<<GB(Ee, T), T, 0, stream>>>(z, nei, nei + Ee, Ee, 100, 1, accl);

    // output layer: txm = scatter(xm); out = xm@W3_0 + txm@W3_1 + b3
    scatter2_kernel<<<GB((long long)Ee * 50, T), T, 0, stream>>>(xm, src, dst, nrm, txm, Ee, 100);
    out3_kernel<<<GB(Nn, T), T, 0, stream>>>(xm, txm, W3_0, W3_1, b3, (float*)d_out, Nn, 100);

    finalize_kernel<<<1, 32, 0, stream>>>(accl, c1, c2, (float*)d_out + Nn, Ee);
}
